// MultiHeadAttention_2173253451830
// MI455X (gfx1250) — matrix operand order repaired and checked by reading, not compiled, not run on hardware
//
#include <hip/hip_runtime.h>
#include <hip/hip_bf16.h>

typedef __attribute__((ext_vector_type(16))) _Float16 v16h;
typedef __attribute__((ext_vector_type(8)))  _Float16 v8h;
typedef __attribute__((ext_vector_type(8)))  float    v8f;

#define DM 1024
#define NH 16
#define DK 64
#define SS 2048
#define BB 2

union AFrag { v16h v; _Float16 e[16]; };
union CFrag { v8f  v; float    e[8];  };

// A operand: 16xK row-major tile at `base` (leading dim ldm halfs), k-chunk kb (mult of 32).
// ISA layout: lane(0-15)=row M, e[0..7] <-> K=kb+8h.., e[8..15] <-> K=kb+16+8h.. (h=lane>>4)
template <typename PT>
__device__ __forceinline__ v16h load_a16(PT base, int lane, int ldm, int kb) {
  int m = lane & 15, h = lane >> 4;
  AFrag a;
  *(v8h*)&a.e[0] = *(const v8h*)(base + m * ldm + kb + 8 * h);
  *(v8h*)&a.e[8] = *(const v8h*)(base + m * ldm + kb + 16 + 8 * h);
  return a.v;
}

// B operand: KxN row-major at `base` (leading dim ldn halfs): lane L holds column nb+(L&15),
// e[0..7] <-> K=kb+8h.., e[8..15] <-> K=kb+16+8h.. (h=L>>4)
__device__ __forceinline__ v16h load_b16(const _Float16* __restrict__ base, int lane,
                                         int ldn, int kb, int nb) {
  int n = lane & 15, h = lane >> 4;
  AFrag b;
#pragma unroll
  for (int s = 0; s < 16; ++s) {
    int k = kb + (s < 8 ? 8 * h + s : 16 + 8 * h + (s - 8));
    b.e[s] = base[(size_t)k * ldn + nb + n];
  }
  return b.v;
}

__device__ __forceinline__ v8f wmma_f16(v16h a, v16h b, v8f c) {
  return __builtin_amdgcn_wmma_f32_16x16x32_f16(false, a, false, b, (short)0, c, false, false);
}

// ---------------- elementwise converts ----------------
__global__ void cvt_kernel(const float* __restrict__ in, _Float16* __restrict__ out, int n) {
  int i = blockIdx.x * blockDim.x + threadIdx.x;
  if (i < n) out[i] = (_Float16)in[i];
}

// wt[k*1024+n] = (f16) w[n*1024+k]
__global__ void twcvt_kernel(const float* __restrict__ w, _Float16* __restrict__ wt) {
  int i = blockIdx.x * blockDim.x + threadIdx.x;
  int kk = i >> 10, n = i & 1023;
  wt[i] = (_Float16)w[n * DM + kk];
}

// ---------------- QKV projection: y = x @ W^T + b, scattered into head layouts ----------------
// Per-wave tile: 32 rows x 64 cols (2 A frags x 4 B frags -> 8 WMMAs per k-step).
// mode 0: Q -> [b,h,s,d] * scale ; mode 1: K -> [b,h,d,s] ; mode 2: V -> [b,h,s,d]
__global__ void proj_kernel(const _Float16* __restrict__ x,
                            const _Float16* __restrict__ wt,
                            const float* __restrict__ bias,
                            _Float16* __restrict__ out,
                            int mode, float scale) {
  int lane = threadIdx.x & 31, wave = threadIdx.x >> 5;
  int row0 = blockIdx.x * 32;
  int col0 = blockIdx.y * 256 + wave * 64;
  CFrag c[2][4];
#pragma unroll
  for (int mt = 0; mt < 2; ++mt)
#pragma unroll
    for (int t = 0; t < 4; ++t)
#pragma unroll
      for (int r = 0; r < 8; ++r) c[mt][t].e[r] = 0.f;

  const _Float16* abase0 = x + (size_t)row0 * DM;
  const _Float16* abase1 = x + (size_t)(row0 + 16) * DM;
  for (int kb = 0; kb < DM; kb += 32) {
    v16h a0 = load_a16(abase0, lane, DM, kb);
    v16h a1 = load_a16(abase1, lane, DM, kb);
    if (kb + 32 < DM) {
      __builtin_prefetch(abase0 + (lane & 15) * DM + kb + 32, 0, 1);
      __builtin_prefetch(wt + (size_t)(kb + 32 + lane) * DM + col0, 0, 1);
    }
#pragma unroll
    for (int t = 0; t < 4; ++t) {
      v16h b = load_b16(wt, lane, DM, kb, col0 + t * 16);
      c[0][t].v = wmma_f16(a0, b, c[0][t].v);
      c[1][t].v = wmma_f16(a1, b, c[1][t].v);
    }
  }

  int mh = lane >> 4, n0 = lane & 15;
#pragma unroll
  for (int mt = 0; mt < 2; ++mt) {
#pragma unroll
    for (int t = 0; t < 4; ++t) {
      int col = col0 + t * 16 + n0;
      float bv = bias[col];
      int hh = col >> 6, d = col & (DK - 1);
#pragma unroll
      for (int r = 0; r < 8; ++r) {
        int row = row0 + mt * 16 + r + 8 * mh;
        int b = row >> 11, s = row & (SS - 1);
        float vv = (c[mt][t].e[r] + bv) * scale;
        size_t addr;
        if (mode == 1) addr = (((size_t)(b * NH + hh)) * DK + d) * SS + s;
        else           addr = (((size_t)(b * NH + hh)) * SS + s) * DK + d;
        out[addr] = (_Float16)vv;
      }
    }
  }
}

// ---------------- attention: per (b,h) and 16-query block ----------------
// LDS: sc[16][2048] f32 scores (128KB), pr[16][2048] f16 probs (64KB), reduction scratch
__global__ void attn_kernel(const _Float16* __restrict__ qh,   // [b,h,s,d] (pre-scaled)
                            const _Float16* __restrict__ kht,  // [b,h,d,s]
                            const _Float16* __restrict__ vh,   // [b,h,s,d]
                            _Float16* __restrict__ ao) {       // [b,s,h*d]
  extern __shared__ char smem[];
  float*    sc  = (float*)smem;                              // 16*2048 f32
  _Float16* pr  = (_Float16*)(smem + 16 * SS * 4);           // 16*2048 f16
  float*    red = (float*)(smem + 16 * SS * 4 + 16 * SS * 2);// 16*8
  float*    rmx = red + 128;                                 // 16
  float*    rnv = rmx + 16;                                  // 16

  int tid = threadIdx.x;
  int lane = tid & 31, wave = tid >> 5;
  int qs0 = blockIdx.x * 16;
  int bh  = blockIdx.y;

  const _Float16* qbase = qh  + ((size_t)bh * SS + qs0) * DK;
  const _Float16* kbase = kht + (size_t)bh * DK * SS;
  const _Float16* vbase = vh  + (size_t)bh * SS * DK;

  // Phase A: scores strip 16 x 2048; Q fragments are loop-invariant -> hoisted
  v16h aq0 = load_a16(qbase, lane, DK, 0);
  v16h aq1 = load_a16(qbase, lane, DK, 32);
  for (int i = 0; i < 32; ++i) {
    int nb = (wave * 32 + i) * 16;
    CFrag c;
#pragma unroll
    for (int r = 0; r < 8; ++r) c.e[r] = 0.f;
    v16h b0 = load_b16(kbase, lane, SS, 0, nb);
    v16h b1 = load_b16(kbase, lane, SS, 32, nb);
    c.v = wmma_f16(aq0, b0, c.v);
    c.v = wmma_f16(aq1, b1, c.v);
    int mh = lane >> 4, n0 = lane & 15;
#pragma unroll
    for (int r = 0; r < 8; ++r)
      sc[(r + 8 * mh) * SS + nb + n0] = c.e[r];
  }
  __syncthreads();

  // Row softmax: 8 threads per row, 256 cols each
  int row = tid >> 3, j = tid & 7;
  {
    const float* p = sc + row * SS + j * 256;
    float lm = -1e30f;
    for (int cidx = 0; cidx < 256; ++cidx) lm = fmaxf(lm, p[cidx]);
    red[row * 8 + j] = lm;
  }
  __syncthreads();
  if (j == 0) {
    float m = red[row * 8];
    for (int t = 1; t < 8; ++t) m = fmaxf(m, red[row * 8 + t]);
    rmx[row] = m;
  }
  __syncthreads();
  {
    float m = rmx[row], ls = 0.f;
    const float* p = sc + row * SS + j * 256;
    _Float16* q = pr + row * SS + j * 256;
    for (int cidx = 0; cidx < 256; ++cidx) {
      float e = __expf(p[cidx] - m);
      q[cidx] = (_Float16)e;
      ls += e;
    }
    red[row * 8 + j] = ls;
  }
  __syncthreads();
  if (j == 0) {
    float s = 0.f;
    for (int t = 0; t < 8; ++t) s += red[row * 8 + t];
    rnv[row] = 1.0f / s;
  }
  __syncthreads();

  // Phase C: out(16 x 64) = P(16 x 2048) @ V(2048 x 64); wave w owns d-tile w
  int dw = wave * 16;
  CFrag c;
#pragma unroll
  for (int r = 0; r < 8; ++r) c.e[r] = 0.f;
  for (int kc = 0; kc < SS; kc += 32) {
    v16h a = load_a16(pr, lane, SS, kc);     // from LDS (ds_read_b128)
    v16h b = load_b16(vbase, lane, DK, kc, dw);
    if (kc + 32 < SS)
      __builtin_prefetch(vbase + (size_t)(kc + 32 + lane) * DK + dw, 0, 1);
    c.v = wmma_f16(a, b, c.v);
  }
  int mh = lane >> 4, n0 = lane & 15;
  int b = bh >> 4, hh = bh & 15;
#pragma unroll
  for (int r = 0; r < 8; ++r) {
    int m = r + 8 * mh;
    float vv = c.e[r] * rnv[m];
    ao[((size_t)b * SS + qs0 + m) * DM + hh * DK + dw + n0] = (_Float16)vv;
  }
}

// ---------------- output projection: f32 result ----------------
// Per-wave tile: 32 rows x 64 cols, same blocking as proj_kernel.
__global__ void oproj_kernel(const _Float16* __restrict__ x,
                             const _Float16* __restrict__ wt,
                             const float* __restrict__ bias,
                             float* __restrict__ out) {
  int lane = threadIdx.x & 31, wave = threadIdx.x >> 5;
  int row0 = blockIdx.x * 32;
  int col0 = blockIdx.y * 256 + wave * 64;
  CFrag c[2][4];
#pragma unroll
  for (int mt = 0; mt < 2; ++mt)
#pragma unroll
    for (int t = 0; t < 4; ++t)
#pragma unroll
      for (int r = 0; r < 8; ++r) c[mt][t].e[r] = 0.f;

  const _Float16* abase0 = x + (size_t)row0 * DM;
  const _Float16* abase1 = x + (size_t)(row0 + 16) * DM;
  for (int kb = 0; kb < DM; kb += 32) {
    v16h a0 = load_a16(abase0, lane, DM, kb);
    v16h a1 = load_a16(abase1, lane, DM, kb);
    if (kb + 32 < DM) {
      __builtin_prefetch(abase0 + (lane & 15) * DM + kb + 32, 0, 1);
      __builtin_prefetch(wt + (size_t)(kb + 32 + lane) * DM + col0, 0, 1);
    }
#pragma unroll
    for (int t = 0; t < 4; ++t) {
      v16h b = load_b16(wt, lane, DM, kb, col0 + t * 16);
      c[0][t].v = wmma_f16(a0, b, c[0][t].v);
      c[1][t].v = wmma_f16(a1, b, c[1][t].v);
    }
  }
  int mh = lane >> 4, n0 = lane & 15;
#pragma unroll
  for (int mt = 0; mt < 2; ++mt) {
#pragma unroll
    for (int t = 0; t < 4; ++t) {
      int col = col0 + t * 16 + n0;
      float bv = bias[col];
#pragma unroll
      for (int r = 0; r < 8; ++r) {
        int row = row0 + mt * 16 + r + 8 * mh;
        out[(size_t)row * DM + col] = c[mt][t].e[r] + bv;
      }
    }
  }
}

extern "C" void kernel_launch(void* const* d_in, const int* in_sizes, int n_in,
                              void* d_out, int out_size, void* d_ws, size_t ws_size,
                              hipStream_t stream) {
  const float* q  = (const float*)d_in[0];
  const float* k  = (const float*)d_in[1];
  const float* v  = (const float*)d_in[2];
  const float* wq = (const float*)d_in[3];
  const float* bq = (const float*)d_in[4];
  const float* wk = (const float*)d_in[5];
  const float* bk = (const float*)d_in[6];
  const float* wv = (const float*)d_in[7];
  const float* bv = (const float*)d_in[8];
  const float* wo = (const float*)d_in[9];
  const float* bo = (const float*)d_in[10];
  float* out = (float*)d_out;

  const size_t MN = (size_t)BB * SS * DM;  // 4,194,304
  const size_t WN = (size_t)DM * DM;       // 1,048,576

  _Float16* ws  = (_Float16*)d_ws;
  _Float16* qf  = ws;
  _Float16* kf  = qf  + MN;
  _Float16* vf  = kf  + MN;
  _Float16* wqt = vf  + MN;
  _Float16* wkt = wqt + WN;
  _Float16* wvt = wkt + WN;
  _Float16* wot = wvt + WN;
  _Float16* Qh  = wot + WN;
  _Float16* Kht = Qh  + MN;
  _Float16* Vh  = Kht + MN;
  _Float16* Ao  = Vh  + MN;   // total ~67 MB of f16 scratch

  // 1) converts
  cvt_kernel<<<dim3((unsigned)(MN / 256)), 256, 0, stream>>>(q, qf, (int)MN);
  cvt_kernel<<<dim3((unsigned)(MN / 256)), 256, 0, stream>>>(k, kf, (int)MN);
  cvt_kernel<<<dim3((unsigned)(MN / 256)), 256, 0, stream>>>(v, vf, (int)MN);
  twcvt_kernel<<<dim3((unsigned)(WN / 256)), 256, 0, stream>>>(wq, wqt);
  twcvt_kernel<<<dim3((unsigned)(WN / 256)), 256, 0, stream>>>(wk, wkt);
  twcvt_kernel<<<dim3((unsigned)(WN / 256)), 256, 0, stream>>>(wv, wvt);
  twcvt_kernel<<<dim3((unsigned)(WN / 256)), 256, 0, stream>>>(wo, wot);

  // 2) projections (1/sqrt(dk)=0.125 folded into Q); per-wave 32x64 tiles
  proj_kernel<<<dim3(128, 4), 128, 0, stream>>>(qf, wqt, bq, Qh, 0, 0.125f);
  proj_kernel<<<dim3(128, 4), 128, 0, stream>>>(kf, wkt, bk, Kht, 1, 1.0f);
  proj_kernel<<<dim3(128, 4), 128, 0, stream>>>(vf, wvt, bv, Vh, 2, 1.0f);

  // 3) attention, ~192KB dynamic LDS per workgroup (CDNA5 allows up to 320KB)
  size_t smem = (size_t)16 * SS * 4 + (size_t)16 * SS * 2 + (128 + 16 + 16) * 4;
  attn_kernel<<<dim3(SS / 16, BB * NH), 128, smem, stream>>>(Qh, Kht, Vh, Ao);

  // 4) output projection -> f32
  oproj_kernel<<<dim3(128, 4), 128, 0, stream>>>(Ao, wot, bo, out);
}
